// Attention_14912126451817
// MI455X (gfx1250) — matrix operand order repaired and checked by reading, not compiled, not run on hardware
//
#include <hip/hip_runtime.h>

// Problem constants (from reference)
#define B_   4
#define S_   2048
#define D_   2048
#define H_   16
#define KVH_ 8
#define HD_  128
#define M_   (B_ * S_)        // 8192 rows
#define QN_  (H_ * HD_)       // 2048
#define KN_  (KVH_ * HD_)     // 1024

typedef __attribute__((ext_vector_type(16))) __bf16 v16bf;
typedef __attribute__((ext_vector_type(8)))  float  v8f;
typedef __attribute__((ext_vector_type(4)))  unsigned int u32x4;
typedef __attribute__((ext_vector_type(8)))  int i32x8;
typedef __attribute__((ext_vector_type(4)))  int i32x4;

#if defined(__AMDGCN__) && __has_builtin(__builtin_amdgcn_tensor_load_to_lds)
#define HAVE_TDM 1
#else
#define HAVE_TDM 0
#endif

union FragU { v16bf v; uint4 u[2]; };

__device__ __forceinline__ float bf2f(unsigned short h) {
    unsigned int u = ((unsigned int)h) << 16;
    float f; __builtin_memcpy(&f, &u, 4); return f;
}
__device__ __forceinline__ unsigned short f2bf(float f) {
    unsigned int u; __builtin_memcpy(&u, &f, 4);
    u = (u + 0x7FFFu + ((u >> 16) & 1u)) >> 16;   // round-to-nearest-even
    return (unsigned short)u;
}

__device__ __forceinline__ v8f wmma_bf16(const FragU& a, const FragU& b, v8f c) {
    return __builtin_amdgcn_wmma_f32_16x16x32_bf16(false, a.v, false, b.v,
                                                   (short)0, c, false, false);
}

// Cross-lane reductions confined to a 16-lane half (C-fragment rows live there)
__device__ __forceinline__ float hmax16(float v) {
    v = fmaxf(v, __shfl_xor(v, 1));
    v = fmaxf(v, __shfl_xor(v, 2));
    v = fmaxf(v, __shfl_xor(v, 4));
    v = fmaxf(v, __shfl_xor(v, 8));
    return v;
}
__device__ __forceinline__ float hsum16(float v) {
    v += __shfl_xor(v, 1);
    v += __shfl_xor(v, 2);
    v += __shfl_xor(v, 4);
    v += __shfl_xor(v, 8);
    return v;
}

#if HAVE_TDM
// Issue one TDM 2-D tile load: 128 rows x 64 bytes (32 bf16) into contiguous
// LDS at lds_off.  Row stride = row_stride_bytes.  data_size = 8B units.
// D# packing per CDNA5 ISA ch.8: group0 {count,lds,global,type}, group1
// {mask,dsize,dims,strides}, groups 2/3 zero (pure 2-D tile).
__device__ __forceinline__ void tdm_load_tile(unsigned lds_off, const void* gptr,
                                              long long row_stride_bytes) {
    unsigned long long ga = (unsigned long long)gptr;
    long long stride_u = row_stride_bytes >> 3;         // 8-byte units

    u32x4 g0;
    g0[0] = 1u;                                         // count=1, user mode
    g0[1] = lds_off;                                    // lds_addr
    g0[2] = (unsigned)ga;                               // global_addr[31:0]
    g0[3] = (unsigned)((ga >> 32) & 0x01FFFFFFu)        // global_addr[56:32]
          | (2u << 30);                                 // type = image

    i32x8 g1;
    g1[0] = 3 << 16;                                    // data_size = 8B
    g1[1] = 0;                                          // tensor_dim0 lo16<<16 (dim0=1<<30)
    g1[2] = 0x4000;                                     // tensor_dim0 hi16; dim1 lo16=0
    g1[3] = 0x10 | (8 << 16);                           // dim1 hi16 (1<<20); tile_dim0=8
    g1[4] = 128;                                        // tile_dim1=128, tile_dim2=0
    g1[5] = (int)(unsigned)(stride_u & 0xFFFFFFFFll);   // dim0_stride[31:0]
    g1[6] = (int)(unsigned)((stride_u >> 32) & 0xFFFFll); // dim0_stride[47:32]
    g1[7] = 0;

    i32x4 gz = {0, 0, 0, 0};
#if __clang_major__ >= 23
    i32x8 gz8 = {0, 0, 0, 0, 0, 0, 0, 0};
    __builtin_amdgcn_tensor_load_to_lds(g0, g1, gz, gz, gz8, 0);
#else
    __builtin_amdgcn_tensor_load_to_lds(g0, g1, gz, gz, 0);
#endif
}
#endif

// ---------------------------------------------------------------- elementwise
__global__ void cvt_f32_bf16(const float* __restrict__ in,
                             unsigned short* __restrict__ out, int n) {
    int i = blockIdx.x * blockDim.x + threadIdx.x;
    if (i < n) out[i] = f2bf(in[i]);
}

// in: (Kd x Nd) fp32 row-major  ->  out: (Nd x Kd) bf16 row-major (W^T)
__global__ void transpose_cvt(const float* __restrict__ in,
                              unsigned short* __restrict__ out,
                              int Kd, int Nd, int total) {
    int i = blockIdx.x * blockDim.x + threadIdx.x;
    if (i >= total) return;
    int n = i / Kd, k = i - n * Kd;
    out[i] = f2bf(in[(size_t)k * Nd + n]);
}

// RoPE in-place on bf16 tensor of shape (B,S,heads,HD); pairs (2i,2i+1)
__global__ void rope_bf16(unsigned short* __restrict__ t,
                          const float* __restrict__ cs,
                          const float* __restrict__ sn,
                          int heads, int total) {
    int idx = blockIdx.x * blockDim.x + threadIdx.x;
    if (idx >= total) return;
    int i  = idx % (HD_ / 2);
    int tm = idx / (HD_ / 2);
    int hh = tm % heads; tm /= heads;
    int s  = tm % S_;
    int b  = tm / S_;
    size_t base = (((size_t)b * S_ + s) * heads + hh) * HD_ + 2 * i;
    float c  = cs[s * (HD_ / 2) + i];
    float sv = sn[s * (HD_ / 2) + i];
    float t0 = bf2f(t[base]), t1 = bf2f(t[base + 1]);
    t[base]     = f2bf(t0 * c - t1 * sv);
    t[base + 1] = f2bf(t0 * sv + t1 * c);
}

// V (B,S,KVH,HD) bf16 -> V^T (B,KVH,HD,S) bf16
__global__ void transpose_v(const unsigned short* __restrict__ v,
                            unsigned short* __restrict__ vt, int total) {
    int i = blockIdx.x * blockDim.x + threadIdx.x;
    if (i >= total) return;
    int s  = i % S_;
    int d  = (i / S_) % HD_;
    int hk = (i / (S_ * HD_)) % KVH_;
    int b  = i / (S_ * HD_ * KVH_);
    vt[i] = v[(((size_t)b * S_ + s) * KVH_ + hk) * HD_ + d];
}

// ---------------------------------------------------------------- WMMA GEMM
// C(M x Nd) = A(M x Kd, bf16 row-major) * Bt(Nd x Kd, bf16 row-major)^T
// Block: 128x128 tile, 8 waves; wave computes 32x64.
// TDM double-buffered: wave 0 streams the next 128x32 A/B tiles into the
// alternate LDS buffer (tensor_load_to_lds, TENSORcnt) while all waves run
// WMMA on the current buffer.  One workgroup barrier per K-step.
__global__ __launch_bounds__(256)
void gemm_wmma(const unsigned short* __restrict__ A,
               const unsigned short* __restrict__ Bt,
               void* __restrict__ Cout, int Nd, int Kd, int c_is_f32) {
    __shared__ alignas(16) unsigned short lA[2][128 * 32];
    __shared__ alignas(16) unsigned short lB[2][128 * 32];

    const int t    = threadIdx.x;
    const int lane = t & 31;
    const int w    = t >> 5;
    const int m0   = blockIdx.y * 128;
    const int n0   = blockIdx.x * 128;
    const int mt   = (w & 3) * 32;
    const int nt   = (w >> 2) * 64;
    const int ml   = lane & 15;
    const int hl   = lane >> 4;
    const int nsteps = Kd >> 5;

    const v8f vz = {0.f, 0.f, 0.f, 0.f, 0.f, 0.f, 0.f, 0.f};
    v8f acc[2][4];
#pragma unroll
    for (int i = 0; i < 2; ++i)
#pragma unroll
        for (int j = 0; j < 4; ++j) acc[i][j] = vz;

#if HAVE_TDM
    const unsigned ldsA = (unsigned)(unsigned long long)(&lA[0][0]);
    const unsigned ldsB = (unsigned)(unsigned long long)(&lB[0][0]);
    const long long strideBytes = (long long)Kd * 2;

    if (w == 0) {   // prologue: stage step 0 into buffer 0
        tdm_load_tile(ldsA, A  + (size_t)m0 * Kd, strideBytes);
        tdm_load_tile(ldsB, Bt + (size_t)n0 * Kd, strideBytes);
    }
    for (int i = 0; i < nsteps; ++i) {
        const int cur = i & 1;
        if (w == 0) __builtin_amdgcn_s_wait_tensorcnt(0);
        __syncthreads();   // buf[cur] ready; buf[cur^1] fully consumed
        if (w == 0 && i + 1 < nsteps) {
            const int k1 = (i + 1) * 32;
            tdm_load_tile(ldsA + (cur ^ 1) * (128 * 64),
                          A  + (size_t)m0 * Kd + k1, strideBytes);
            tdm_load_tile(ldsB + (cur ^ 1) * (128 * 64),
                          Bt + (size_t)n0 * Kd + k1, strideBytes);
        }
#else
    for (int i = 0; i < nsteps; ++i) {
        const int cur = 0;
        const int k0 = i * 32;
#pragma unroll
        for (int rep = 0; rep < 2; ++rep) {
            int c    = t + rep * 256;      // 512 chunks of 16B per tile
            int row  = c >> 2;
            int colb = (c & 3) * 8;
            ((uint4*)lA[0])[c] = *(const uint4*)(A  + (size_t)(m0 + row) * Kd + k0 + colb);
            ((uint4*)lB[0])[c] = *(const uint4*)(Bt + (size_t)(n0 + row) * Kd + k0 + colb);
        }
        __syncthreads();
#endif
        FragU a[2], b[4];
#pragma unroll
        for (int ii = 0; ii < 2; ++ii) {
            const unsigned short* p = &lA[cur][(mt + ii * 16 + ml) * 32 + hl * 8];
            a[ii].u[0] = *(const uint4*)p;
            a[ii].u[1] = *(const uint4*)(p + 16);
        }
#pragma unroll
        for (int j = 0; j < 4; ++j) {
            const unsigned short* p = &lB[cur][(nt + j * 16 + ml) * 32 + hl * 8];
            b[j].u[0] = *(const uint4*)p;
            b[j].u[1] = *(const uint4*)(p + 16);
        }
#pragma unroll
        for (int ii = 0; ii < 2; ++ii)
#pragma unroll
            for (int j = 0; j < 4; ++j)
                acc[ii][j] = wmma_bf16(a[ii], b[j], acc[ii][j]);
#if !HAVE_TDM
        __syncthreads();
#endif
    }

    if (c_is_f32) {
        float* C = (float*)Cout;
#pragma unroll
        for (int i = 0; i < 2; ++i)
#pragma unroll
            for (int j = 0; j < 4; ++j) {
                int n = n0 + nt + j * 16 + ml;
#pragma unroll
                for (int r = 0; r < 8; ++r) {
                    int m = m0 + mt + i * 16 + r + 8 * hl;
                    C[(size_t)m * Nd + n] = acc[i][j][r];
                }
            }
    } else {
        unsigned short* C = (unsigned short*)Cout;
#pragma unroll
        for (int i = 0; i < 2; ++i)
#pragma unroll
            for (int j = 0; j < 4; ++j) {
                int n = n0 + nt + j * 16 + ml;
#pragma unroll
                for (int r = 0; r < 8; ++r) {
                    int m = m0 + mt + i * 16 + r + 8 * hl;
                    C[(size_t)m * Nd + n] = f2bf(acc[i][j][r]);
                }
            }
    }
}

// ---------------------------------------------------------------- attention
// Q (B,S,H,HD) bf16 rope'd; K (B,S,KVH,HD) bf16 rope'd; VT (B,KVH,HD,S) bf16.
// Block 256 = 8 waves, each wave one 16-query tile. Causal flash softmax.
__global__ __launch_bounds__(256)
void attn_fwd(const unsigned short* __restrict__ Q,
              const unsigned short* __restrict__ K,
              const unsigned short* __restrict__ VT,
              unsigned short* __restrict__ O) {
    __shared__ alignas(16) unsigned short lP[8 * 512];   // 16x32 P tile / wave

    const int t     = threadIdx.x;
    const int lane  = t & 31;
    const int w     = t >> 5;
    const int b     = blockIdx.z;
    const int h     = blockIdx.y;
    const int kvh   = h >> 1;                 // jnp.repeat(k, 2, axis=2)
    const int qbase = blockIdx.x * 128 + w * 16;
    const int ml    = lane & 15;
    const int hl    = lane >> 4;
    const float scale = 0.08838834764831845f; // 1/sqrt(128)

    // Resident Q fragments: 16 x 128, four K=32 slices (A layout)
    FragU qa[4];
    const unsigned short* qrow =
        Q + (((size_t)b * S_ + qbase + ml) * H_ + h) * HD_;
#pragma unroll
    for (int dc = 0; dc < 4; ++dc) {
        qa[dc].u[0] = *(const uint4*)(qrow + dc * 32 + hl * 8);
        qa[dc].u[1] = *(const uint4*)(qrow + dc * 32 + 16 + hl * 8);
    }

    const v8f vz = {0.f, 0.f, 0.f, 0.f, 0.f, 0.f, 0.f, 0.f};
    v8f o[8];
    float mi[8], li[8];
#pragma unroll
    for (int j = 0; j < 8; ++j) o[j] = vz;
#pragma unroll
    for (int r = 0; r < 8; ++r) { mi[r] = -__builtin_inff(); li[r] = 0.f; }

    const unsigned short* kbase = K + ((size_t)b * S_ * KVH_ + kvh) * HD_;
    const unsigned short* vbase = VT + ((size_t)b * KVH_ + kvh) * (size_t)HD_ * S_;
    unsigned short* lp = &lP[w * 512];

    const int nkv = qbase + 16;               // causal extent for this tile
    for (int kb = 0; kb < nkv; kb += 32) {
        // scores: two 16x16 tiles over 32 keys, K-dim = HD (4 WMMAs each)
        v8f s0 = vz, s1 = vz;
#pragma unroll
        for (int g = 0; g < 2; ++g) {
            const int key = kb + g * 16 + ml;
            const unsigned short* kr = kbase + (size_t)key * (KVH_ * HD_);
            FragU kf[4];
#pragma unroll
            for (int dc = 0; dc < 4; ++dc) {
                kf[dc].u[0] = *(const uint4*)(kr + dc * 32 + hl * 8);
                kf[dc].u[1] = *(const uint4*)(kr + dc * 32 + 16 + hl * 8);
            }
            if (g == 0) {
#pragma unroll
                for (int dc = 0; dc < 4; ++dc) s0 = wmma_bf16(qa[dc], kf[dc], s0);
            } else {
#pragma unroll
                for (int dc = 0; dc < 4; ++dc) s1 = wmma_bf16(qa[dc], kf[dc], s1);
            }
        }

        // online softmax per row (rows r -> query qbase + r + 8*hl)
#pragma unroll
        for (int r = 0; r < 8; ++r) {
            const int qi = qbase + r + 8 * hl;
            float v0 = s0[r] * scale;
            float v1 = s1[r] * scale;
            if (kb + ml > qi)       v0 = -__builtin_inff();
            if (kb + 16 + ml > qi)  v1 = -__builtin_inff();
            float cm = hmax16(fmaxf(v0, v1));
            float nm = fmaxf(mi[r], cm);
            float al = __expf(mi[r] - nm);
            mi[r] = nm;
            float p0 = __expf(v0 - nm);
            float p1 = __expf(v1 - nm);
            li[r] = li[r] * al + hsum16(p0 + p1);
#pragma unroll
            for (int j = 0; j < 8; ++j) o[j][r] *= al;
            // stage P (C layout -> LDS row-major 16x32)
            lp[(r + 8 * hl) * 32 + ml]      = f2bf(p0);
            lp[(r + 8 * hl) * 32 + 16 + ml] = f2bf(p1);
        }
        asm volatile("s_wait_dscnt 0" ::: "memory");

        // P as A-fragment (16x32)
        FragU pf;
        pf.u[0] = *(const uint4*)(lp + ml * 32 + hl * 8);
        pf.u[1] = *(const uint4*)(lp + ml * 32 + 16 + hl * 8);

        // O += P @ V  (8 d-tiles of 16, K-dim = 32 keys, V^T contiguous)
#pragma unroll
        for (int j = 0; j < 8; ++j) {
            const unsigned short* vr =
                vbase + (size_t)(j * 16 + ml) * S_ + kb + hl * 8;
            FragU vf;
            vf.u[0] = *(const uint4*)vr;
            vf.u[1] = *(const uint4*)(vr + 16);
            o[j] = wmma_bf16(pf, vf, o[j]);
        }
    }

    // normalize and store (B,S,H,HD) bf16
    float inv[8];
#pragma unroll
    for (int r = 0; r < 8; ++r) inv[r] = 1.f / li[r];
#pragma unroll
    for (int j = 0; j < 8; ++j)
#pragma unroll
        for (int r = 0; r < 8; ++r) {
            int m = r + 8 * hl;
            unsigned short* orow =
                O + (((size_t)b * S_ + qbase + m) * H_ + h) * HD_;
            orow[j * 16 + ml] = f2bf(o[j][r] * inv[r]);
        }
}

// ---------------------------------------------------------------- launcher
extern "C" void kernel_launch(void* const* d_in, const int* in_sizes, int n_in,
                              void* d_out, int out_size, void* d_ws, size_t ws_size,
                              hipStream_t stream) {
    const float* x    = (const float*)d_in[0];
    const float* wq   = (const float*)d_in[1];
    const float* wk   = (const float*)d_in[2];
    const float* wv   = (const float*)d_in[3];
    const float* wo   = (const float*)d_in[4];
    const float* pcos = (const float*)d_in[5];
    const float* psin = (const float*)d_in[6];
    float* out = (float*)d_out;

    char* ws = (char*)d_ws;
    size_t off = 0;
    auto carve = [&](size_t bytes) -> void* {
        void* p = ws + off;
        off += (bytes + 255) & ~(size_t)255;
        return p;
    };
    unsigned short* xb  = (unsigned short*)carve((size_t)M_ * D_ * 2);
    unsigned short* wqT = (unsigned short*)carve((size_t)QN_ * D_ * 2);
    unsigned short* wkT = (unsigned short*)carve((size_t)KN_ * D_ * 2);
    unsigned short* wvT = (unsigned short*)carve((size_t)KN_ * D_ * 2);
    unsigned short* woT = (unsigned short*)carve((size_t)D_ * QN_ * 2);
    unsigned short* qb  = (unsigned short*)carve((size_t)M_ * QN_ * 2);
    unsigned short* kb  = (unsigned short*)carve((size_t)M_ * KN_ * 2);
    unsigned short* vb  = (unsigned short*)carve((size_t)M_ * KN_ * 2);
    unsigned short* vtb = (unsigned short*)carve((size_t)M_ * KN_ * 2);
    unsigned short* ab  = (unsigned short*)carve((size_t)M_ * QN_ * 2);

    const int TH = 256;
    auto blocks = [&](long long n) { return (unsigned)((n + TH - 1) / TH); };

    // 1) precision conversion / weight transposes
    cvt_f32_bf16<<<blocks((long long)M_ * D_), TH, 0, stream>>>(x, xb, M_ * D_);
    transpose_cvt<<<blocks((long long)QN_ * D_), TH, 0, stream>>>(wq, wqT, D_, QN_, QN_ * D_);
    transpose_cvt<<<blocks((long long)KN_ * D_), TH, 0, stream>>>(wk, wkT, D_, KN_, KN_ * D_);
    transpose_cvt<<<blocks((long long)KN_ * D_), TH, 0, stream>>>(wv, wvT, D_, KN_, KN_ * D_);
    transpose_cvt<<<blocks((long long)D_ * QN_), TH, 0, stream>>>(wo, woT, QN_, D_, D_ * QN_);

    // 2) projections
    gemm_wmma<<<dim3(QN_ / 128, M_ / 128), TH, 0, stream>>>(xb, wqT, qb, QN_, D_, 0);
    gemm_wmma<<<dim3(KN_ / 128, M_ / 128), TH, 0, stream>>>(xb, wkT, kb, KN_, D_, 0);
    gemm_wmma<<<dim3(KN_ / 128, M_ / 128), TH, 0, stream>>>(xb, wvT, vb, KN_, D_, 0);

    // 3) RoPE on Q and K
    rope_bf16<<<blocks((long long)B_ * S_ * H_ * HD_ / 2), TH, 0, stream>>>(
        qb, pcos, psin, H_, B_ * S_ * H_ * HD_ / 2);
    rope_bf16<<<blocks((long long)B_ * S_ * KVH_ * HD_ / 2), TH, 0, stream>>>(
        kb, pcos, psin, KVH_, B_ * S_ * KVH_ * HD_ / 2);

    // 4) V^T for contiguous P@V fragments
    transpose_v<<<blocks((long long)B_ * KVH_ * HD_ * S_), TH, 0, stream>>>(
        vb, vtb, B_ * KVH_ * HD_ * S_);

    // 5) causal flash attention
    attn_fwd<<<dim3(S_ / 128, H_, B_), TH, 0, stream>>>(qb, kb, vtb, ab);

    // 6) output projection (fp32 out)
    gemm_wmma<<<dim3(D_ / 128, M_ / 128), TH, 0, stream>>>(ab, woT, out, D_, QN_, 1);

    (void)in_sizes; (void)n_in; (void)out_size; (void)ws_size;
}
